// ESNReservoir_49830210568446
// MI455X (gfx1250) — matrix operand order repaired and checked by reading, not compiled, not run on hardware
//
#include <hip/hip_runtime.h>

#define R       8192
#define TSTEPS  512
#define KT      256   // 8192 / 32  : number of K tiles
#define MT      512   // 8192 / 16  : number of M (row) tiles

typedef __attribute__((ext_vector_type(16))) __bf16 v16bf;
typedef __attribute__((ext_vector_type(8)))  float  v8f;

__device__ __forceinline__ unsigned short f32_bf16(float f) {
    unsigned int u = __float_as_uint(f);
    u += 0x7FFFu + ((u >> 16) & 1u);     // round-to-nearest-even
    return (unsigned short)(u >> 16);
}

// ---------------------------------------------------------------------------
// Pack W (fp32 row-major 8192x8192) into bf16 WMMA A-fragment layout.
// Fragment element (tile, lane l, halfword j):
//   row = m0*16 + (l & 15)
//   col = k0*32 + ((l < 16) ? 0 : 8) + (j & 7) + ((j >> 3) << 4)
// Also zero-initializes h0.
// ---------------------------------------------------------------------------
__global__ __launch_bounds__(256)
void esn_pack(const float* __restrict__ W, unsigned short* __restrict__ Wp,
              float* __restrict__ h0) {
    long long idx = (long long)blockIdx.x * 256 + threadIdx.x;  // 0 .. 64Mi-1
    int j = (int)(idx & 15);
    int l = (int)((idx >> 4) & 31);
    long long tile = idx >> 9;
    int k0 = (int)(tile & (KT - 1));
    int m0 = (int)(tile >> 8);
    int row = m0 * 16 + (l & 15);
    int col = k0 * 32 + ((l < 16) ? 0 : 8) + (j & 7) + ((j >> 3) << 4);
    Wp[idx] = f32_bf16(W[(long long)row * R + col]);
    if (idx < R) h0[idx] = 0.0f;
}

// ---------------------------------------------------------------------------
// One recurrence step: h_out = tanh(Win @ x_t + W @ h_prev)
// Grid: 256 blocks x 256 threads (8 waves). Block owns 2 row-tiles (32 rows);
// 4 waves split the K dimension per row-tile. W@h via v_wmma_f32_16x16x32_bf16
// with h replicated across all 16 B columns (every D column = W_tile @ h).
// ---------------------------------------------------------------------------
__global__ __launch_bounds__(256)
void esn_step(const unsigned short* __restrict__ Wp,
              const float* __restrict__ Win,
              const float* __restrict__ x,
              const float* __restrict__ h_prev,
              float* __restrict__ h_out,
              int t) {
    __shared__ __align__(32) unsigned short hb[R];   // 16 KB bf16 h
    __shared__ float part[2][4][16];

    for (int i = threadIdx.x; i < R; i += 256)
        hb[i] = f32_bf16(h_prev[i]);
    __syncthreads();

    const int lane  = threadIdx.x & 31;
    const int wave  = threadIdx.x >> 5;
    const int rt    = wave >> 2;                 // local row tile 0..1
    const int ks    = wave & 3;                  // K split 0..3
    const int m0    = blockIdx.x * 2 + rt;       // global row tile 0..511
    const int khalf = (lane < 16) ? 0 : 8;       // B: lanes 0-15 K=0..7,16..23; 16-31 K=8..15,24..31

    const unsigned short* ap = Wp + ((long long)m0 * KT + (long long)ks * 64) * 512
                                  + lane * 16;

    v8f acc = {};
    for (int k0 = 0; k0 < 64; ++k0) {
        v16bf a = *(const v16bf*)(ap);
        const unsigned short* bp = &hb[(ks * 64 + k0) * 32 + khalf];
        uint4 blo = *(const uint4*)(bp);
        uint4 bhi = *(const uint4*)(bp + 16);
        struct { uint4 lo, hi; } bpair = {blo, bhi};
        v16bf b = __builtin_bit_cast(v16bf, bpair);
        acc = __builtin_amdgcn_wmma_f32_16x16x32_bf16(
                  false, a, false, b, (short)0, acc, false, false);
        ap += 512;
    }

    // D layout: VGPR i, lanes 0-15 -> M=i ; lanes 16-31 -> M=i+8 (all N equal)
    if (lane == 0 || lane == 16) {
        int half = (lane >> 4) * 8;
        #pragma unroll
        for (int i = 0; i < 8; ++i)
            part[rt][ks][half + i] = acc[i];
    }
    __syncthreads();

    if (threadIdx.x < 32) {
        int rl  = threadIdx.x;          // 0..31 local row
        int lrt = rl >> 4;
        int ri  = rl & 15;
        float s = part[lrt][0][ri] + part[lrt][1][ri]
                + part[lrt][2][ri] + part[lrt][3][ri];
        int r   = blockIdx.x * 32 + rl;
        float x0 = x[t * 3 + 0], x1 = x[t * 3 + 1], x2 = x[t * 3 + 2];
        s += Win[r * 3 + 0] * x0 + Win[r * 3 + 1] * x1 + Win[r * 3 + 2] * x2;
        h_out[r] = tanhf(s);
    }
}

// ---------------------------------------------------------------------------
// y[t] = Wout_w @ h[t] + Wout_b, one block per timestep.
// ---------------------------------------------------------------------------
__global__ __launch_bounds__(256)
void esn_out(const float* __restrict__ hs,
             const float* __restrict__ Wout,
             const float* __restrict__ bias,
             float* __restrict__ y) {
    __shared__ float s0[256], s1[256], s2[256];
    int t = blockIdx.x;
    const float* h = hs + (long long)t * R;
    float a0 = 0.f, a1 = 0.f, a2 = 0.f;
    for (int r = threadIdx.x; r < R; r += 256) {
        float hv = h[r];
        a0 += Wout[r] * hv;
        a1 += Wout[R + r] * hv;
        a2 += Wout[2 * R + r] * hv;
    }
    s0[threadIdx.x] = a0; s1[threadIdx.x] = a1; s2[threadIdx.x] = a2;
    __syncthreads();
    for (int off = 128; off > 0; off >>= 1) {
        if ((int)threadIdx.x < off) {
            s0[threadIdx.x] += s0[threadIdx.x + off];
            s1[threadIdx.x] += s1[threadIdx.x + off];
            s2[threadIdx.x] += s2[threadIdx.x + off];
        }
        __syncthreads();
    }
    if (threadIdx.x == 0) {
        y[t * 3 + 0] = s0[0] + bias[0];
        y[t * 3 + 1] = s1[0] + bias[1];
        y[t * 3 + 2] = s2[0] + bias[2];
    }
}

extern "C" void kernel_launch(void* const* d_in, const int* in_sizes, int n_in,
                              void* d_out, int out_size, void* d_ws, size_t ws_size,
                              hipStream_t stream) {
    const float* x      = (const float*)d_in[0];   // (1, 512, 3)
    const float* Win    = (const float*)d_in[1];   // (8192, 3)
    const float* W      = (const float*)d_in[2];   // (8192, 8192)
    const float* Wout_w = (const float*)d_in[3];   // (3, 8192)
    const float* Wout_b = (const float*)d_in[4];   // (3,)
    float* out = (float*)d_out;                    // (1, 512, 3)

    // workspace layout: packed bf16 W | h0 (zeros) | h states (512 x 8192 fp32)
    unsigned short* Wp = (unsigned short*)d_ws;
    size_t wp_bytes = (size_t)R * R * sizeof(unsigned short);      // 134 MB
    float* h0 = (float*)((char*)d_ws + wp_bytes);
    float* hs = (float*)((char*)d_ws + wp_bytes + (size_t)R * sizeof(float));

    // 1) pack W to bf16 WMMA-A layout (+ zero h0)
    esn_pack<<<dim3((R * (size_t)R) / 256), dim3(256), 0, stream>>>(W, Wp, h0);

    // 2) 512 sequential recurrence steps (kernel boundary = device-wide sync)
    for (int t = 0; t < TSTEPS; ++t) {
        const float* hp = (t == 0) ? h0 : (hs + (size_t)(t - 1) * R);
        esn_step<<<dim3(256), dim3(256), 0, stream>>>(Wp, Win, x, hp,
                                                      hs + (size_t)t * R, t);
    }

    // 3) all readouts in one pass
    esn_out<<<dim3(TSTEPS), dim3(256), 0, stream>>>(hs, Wout_w, Wout_b, out);
}
